// TGN_39754217292129
// MI455X (gfx1250) — matrix operand order repaired and checked by reading, not compiled, not run on hardware
//
#include <hip/hip_runtime.h>
#include <math.h>
#include <limits.h>

// ---------------------------------------------------------------------------
// TGN memory update for MI455X (gfx1250, wave32, WMMA).
//
// Dims (fixed by reference): N=200000 nodes, E=100000 edges, H=100, R=172,
// T=100.  msg_dim = 2H+R+T = 472.  GEMM K padded: 472 msg + 100 mem + 4 pad
// = 576.  Output columns NC = 400: [0,100)=r (i_r+h_r fused via K),
// [100,200)=z, [200,300)=i_n (mem-K zeroed), [300,400)=h_n (msg-K zeroed).
//
// GEMM kernel: 4 waves / 16-row tile; A tile staged to LDS via the CDNA5
// async global->LDS path (GLOBAL_LOAD_ASYNC_TO_LDS_B128 + s_wait_asynccnt);
// each wave runs v_wmma_f32_16x16x32_f16 over a strided set of col tiles,
// C staged through LDS, fused GRU epilogue.
// ---------------------------------------------------------------------------

typedef __attribute__((ext_vector_type(16))) _Float16      v16h;
typedef __attribute__((ext_vector_type(8)))  float         v8f;
typedef __attribute__((ext_vector_type(4)))  unsigned int  u32x4;

union V16 { v16h h; u32x4 q[2]; };

#define HDIM 100
#define RDIM 172
#define TDIM 100
#define KDIM 576     // 472 msg + 100 mem + 4 pad
#define NCOL 400     // r | z | i_n | h_n

// ---------------------------------------------------------------- init
__global__ void k_init(int* __restrict__ maxt, int* __restrict__ win, int N) {
    int i = blockIdx.x * blockDim.x + threadIdx.x;
    if (i >= N) return;
    maxt[i] = INT_MIN;
    win[i]  = -1;
}

// --------------------------------------------- LastAggregator pass 1: max t
__global__ void k_scatter_maxt(const int* __restrict__ src, const int* __restrict__ dst,
                               const int* __restrict__ t, int* __restrict__ maxt, int E) {
    int g = blockIdx.x * blockDim.x + threadIdx.x;
    if (g >= 2 * E) return;
    int e    = (g < E) ? g : g - E;
    int node = (g < E) ? src[e] : dst[e];
    atomicMax(&maxt[node], t[e]);
}

// ---------------------------------- pass 2: winning edge id (tie -> max gid)
__global__ void k_scatter_win(const int* __restrict__ src, const int* __restrict__ dst,
                              const int* __restrict__ t, const int* __restrict__ maxt,
                              int* __restrict__ win, int E) {
    int g = blockIdx.x * blockDim.x + threadIdx.x;
    if (g >= 2 * E) return;
    int e    = (g < E) ? g : g - E;
    int node = (g < E) ? src[e] : dst[e];
    if (t[e] == maxt[node]) atomicMax(&win[node], g);
}

// ------------------------------------- build packed f16 A matrix [N, KDIM]
// cols [0,100)   mem[self]   (zero if no message)
// cols [100,200) mem[other]  (zero if no message)
// cols [200,372) raw_msg[e]  (zero if no message)
// cols [372,472) cos(t_rel*te_w + te_b)            (zero if no message)
// cols [472,572) mem[self]   (always; feeds the gh part of the fused GEMM)
// cols [572,576) 0 pad
__global__ void k_build_msg(const int* __restrict__ src, const int* __restrict__ dst,
                            const int* __restrict__ t, const float* __restrict__ raw,
                            const float* __restrict__ memory, const int* __restrict__ lu,
                            const float* __restrict__ te_w, const float* __restrict__ te_b,
                            const int* __restrict__ win, _Float16* __restrict__ A,
                            int N, int E) {
    int tid = threadIdx.x;
    int n   = blockIdx.x * 4 + (tid >> 6);      // 4 nodes / 256-thread block
    if (n >= N) return;
    int c0  = tid & 63;

    int w = win[n];
    int e = 0, other = 0;
    if (w >= 0) {
        e = (w < E) ? w : w - E;
        int s = src[e], d = dst[e];
        other = (w < E) ? d : s;                // self == n by construction
    }
    float trel = (w >= 0) ? (float)(t[e] - lu[n]) : 0.0f;

    for (int col = c0; col < KDIM; col += 64) {
        float val;
        if (col >= 572)                 val = 0.0f;
        else if (col >= 472)            val = memory[(size_t)n * HDIM + (col - 472)];
        else if (w < 0)                 val = 0.0f;
        else if (col < 100)             val = memory[(size_t)n * HDIM + col];
        else if (col < 200)             val = memory[(size_t)other * HDIM + (col - 100)];
        else if (col < 372)             val = raw[(size_t)e * RDIM + (col - 200)];
        else {
            int j = col - 372;
            val = cosf(trel * te_w[j] + te_b[j]);
        }
        A[(size_t)n * KDIM + col] = (_Float16)val;
    }
}

// ------------------------------- pack B as Bt[NCOL][KDIM] f16 (column-major B)
__global__ void k_pack_b(const float* __restrict__ w_ih, const float* __restrict__ w_hh,
                         _Float16* __restrict__ Bt) {
    int idx = blockIdx.x * blockDim.x + threadIdx.x;
    if (idx >= NCOL * KDIM) return;
    int nn = idx / KDIM, kk = idx % KDIM;
    float v = 0.0f;
    if (nn < 200) {                                   // r, z: i-part + h-part fused in K
        if (kk < 472)       v = w_ih[(size_t)nn * 472 + kk];
        else if (kk < 572)  v = w_hh[(size_t)nn * 100 + (kk - 472)];
    } else if (nn < 300) {                            // i_n: msg K only
        if (kk < 472)       v = w_ih[(size_t)nn * 472 + kk];
    } else {                                          // h_n: mem K only
        if (kk >= 472 && kk < 572) v = w_hh[(size_t)(nn - 100) * 100 + (kk - 472)];
    }
    Bt[(size_t)nn * KDIM + kk] = (_Float16)v;
}

// ------------------ fused WMMA GEMM + GRU epilogue, 4 waves / 16-row tile
__global__ __launch_bounds__(128)
void k_gemm_gru(const _Float16* __restrict__ A, const _Float16* __restrict__ Bt,
                const float* __restrict__ memory, const float* __restrict__ b_ih,
                const float* __restrict__ b_hh, float* __restrict__ out, int N) {
    __shared__ __align__(16) _Float16 sA[16 * KDIM];    // 18 KB
    __shared__ float gbuf[16 * NCOL];                   // 25.6 KB

    const int tid  = threadIdx.x;
    const int lane = tid & 31;
    const int wave = tid >> 5;
    const int row0 = blockIdx.x * 16;

    // ---- stage 16xKDIM A tile into LDS: 1152 x b128 chunks over 128 threads
    if (row0 + 16 <= N) {
        // CDNA5 async global->LDS DMA (no VGPR round-trip); ASYNCcnt-tracked.
        // Low 32 bits of a flat shared-space address == wave-relative LDS
        // byte offset (ISA 10.2 aperture mapping).
        unsigned lds_base = (unsigned)(size_t)(void*)sA;
        const char* gsrc  = (const char*)(A + (size_t)row0 * KDIM);
        #pragma unroll
        for (int i = 0; i < 9; ++i) {                 // 9 = 1152/128 chunks/thread
            unsigned           loff  = lds_base + (unsigned)(tid + i * 128) * 16u;
            unsigned long long gaddr = (unsigned long long)(gsrc + (size_t)(tid + i * 128) * 16);
            asm volatile("global_load_async_to_lds_b128 %0, %1, off"
                         :: "v"(loff), "v"(gaddr) : "memory");
        }
        asm volatile("s_wait_asynccnt 0x0" ::: "memory");
    } else {
        // partial tail tile: scalar path with zero fill
        u32x4*       sAv = (u32x4*)sA;
        const u32x4* gA  = (const u32x4*)(A + (size_t)row0 * KDIM);
        for (int i = tid; i < 16 * (KDIM / 8); i += 128) {
            int rrow = i / (KDIM / 8);
            u32x4 z = {0u, 0u, 0u, 0u};
            sAv[i] = (row0 + rrow < N) ? gA[i] : z;
        }
    }
    __syncthreads();

    const int r     = lane & 15;        // A row / B column within tile
    const int hi    = lane >> 4;        // half-wave select
    const int khalf = hi * 8;           // A: lanes 16-31 carry K+8 / K+24 halves

    // each wave handles col tiles wave, wave+4, ... (25 tiles total)
    for (int ct = wave; ct < NCOL / 16; ct += 4) {
        v8f acc = {};
        const _Float16* bcol = Bt + (size_t)(ct * 16 + r) * KDIM + khalf;
        for (int kb = 0; kb < KDIM; kb += 32) {
            V16 a, b;
            const u32x4* ap = (const u32x4*)(sA + r * KDIM + kb + khalf);
            a.q[0] = ap[0];                       // halves [kb+khalf   .. +7]
            a.q[1] = ap[2];                       // halves [kb+16+khalf.. +7]
            const u32x4* bp = (const u32x4*)(bcol + kb);
            b.q[0] = bp[0];                       // B col, K halves [kb+khalf..+7]
            b.q[1] = bp[2];
            acc = __builtin_amdgcn_wmma_f32_16x16x32_f16(
                false, a.h, false, b.h, (short)0, acc, false, false);
        }
        // C layout: VGPR v <-> M = v + 8*hi, col = lane&15 (ISA 7.12.2)
        const int crow = hi * 8;
        const int ccol = ct * 16 + r;
        #pragma unroll
        for (int v = 0; v < 8; ++v) gbuf[(crow + v) * NCOL + ccol] = acc[v];
    }
    __syncthreads();

    // GRU epilogue: 16 rows x H cols over 128 threads
    for (int i = tid; i < 16 * HDIM; i += 128) {
        int rr = i / HDIM, c = i % HDIM;
        int node = row0 + rr;
        if (node >= N) continue;
        float gr  = gbuf[rr * NCOL + c]        + b_ih[c]        + b_hh[c];
        float gz  = gbuf[rr * NCOL + 100 + c]  + b_ih[100 + c]  + b_hh[100 + c];
        float gin = gbuf[rr * NCOL + 200 + c]  + b_ih[200 + c];
        float ghn = gbuf[rr * NCOL + 300 + c]  + b_hh[200 + c];
        float rg = 1.0f / (1.0f + __expf(-gr));
        float zg = 1.0f / (1.0f + __expf(-gz));
        float ng = tanhf(gin + rg * ghn);
        float h  = memory[(size_t)node * HDIM + c];
        out[(size_t)node * HDIM + c] = (1.0f - zg) * ng + zg * h;
    }
}

// ----------------------------------------------------- last_update scatter-max
__global__ void k_finalize_lu(const int* __restrict__ lu, const int* __restrict__ maxt,
                              int* __restrict__ lu_out, int N) {
    int i = blockIdx.x * blockDim.x + threadIdx.x;
    if (i >= N) return;
    int m = maxt[i], old = lu[i];
    lu_out[i] = (m == INT_MIN) ? old : (m > old ? m : old);
}

// ---------------------------------------------------------------------------
extern "C" void kernel_launch(void* const* d_in, const int* in_sizes, int n_in,
                              void* d_out, int out_size, void* d_ws, size_t ws_size,
                              hipStream_t stream) {
    const int*   src    = (const int*)d_in[0];
    const int*   dst    = (const int*)d_in[1];
    const int*   t      = (const int*)d_in[2];
    const float* raw    = (const float*)d_in[3];
    const float* memory = (const float*)d_in[4];
    const int*   lu     = (const int*)d_in[5];
    const float* te_w   = (const float*)d_in[6];
    const float* te_b   = (const float*)d_in[7];
    const float* w_ih   = (const float*)d_in[8];
    const float* w_hh   = (const float*)d_in[9];
    const float* b_ih   = (const float*)d_in[10];
    const float* b_hh   = (const float*)d_in[11];

    const int E = in_sizes[0];
    const int N = in_sizes[5];

    float* out    = (float*)d_out;
    int*   lu_out = (int*)(out + (size_t)N * HDIM);

    // workspace carve-up (256B aligned)
    char*  ws  = (char*)d_ws;
    size_t off = 0;
    auto carve = [&](size_t bytes) -> char* {
        char* p = ws + off;
        off = (off + bytes + 255) & ~(size_t)255;
        return p;
    };
    int*      maxt = (int*)carve((size_t)N * sizeof(int));
    int*      win  = (int*)carve((size_t)N * sizeof(int));
    _Float16* Amat = (_Float16*)carve((size_t)N * KDIM * sizeof(_Float16));
    _Float16* Bt   = (_Float16*)carve((size_t)NCOL * KDIM * sizeof(_Float16));
    (void)ws_size; (void)n_in; (void)out_size;

    k_init<<<(N + 255) / 256, 256, 0, stream>>>(maxt, win, N);
    k_scatter_maxt<<<(2 * E + 255) / 256, 256, 0, stream>>>(src, dst, t, maxt, E);
    k_scatter_win<<<(2 * E + 255) / 256, 256, 0, stream>>>(src, dst, t, maxt, win, E);
    k_build_msg<<<(N + 3) / 4, 256, 0, stream>>>(src, dst, t, raw, memory, lu,
                                                 te_w, te_b, win, Amat, N, E);
    k_pack_b<<<(NCOL * KDIM + 255) / 256, 256, 0, stream>>>(w_ih, w_hh, Bt);
    k_gemm_gru<<<(N + 15) / 16, 128, 0, stream>>>(Amat, Bt, memory, b_ih, b_hh, out, N);
    k_finalize_lu<<<(N + 255) / 256, 256, 0, stream>>>(lu, maxt, lu_out, N);
}
